// pooling_network_v1_57406532878691
// MI455X (gfx1250) — matrix operand order repaired and checked by reading, not compiled, not run on hardware
//
#include <hip/hip_runtime.h>
#include <hip/hip_bf16.h>
#include <stdint.h>

// ---------------------------------------------------------------------------
// Problem constants (match reference)
// ---------------------------------------------------------------------------
#define NNODE 50000
#define NEDGE 800000
#define DIN0  1044      // IN_FEAT + WALK
#define DIN0P 1056      // padded to multiple of 32
#define HDIM  256
#define NTPW  8         // N-tiles per wave (16x128 output strip)

static const int KSTAGE[5] = {50000, 25000, 12500, 6250, 3125};

// ---------------------------------------------------------------------------
// Types for WMMA
// ---------------------------------------------------------------------------
typedef __attribute__((ext_vector_type(16))) __bf16 v16bf;
typedef __attribute__((ext_vector_type(8)))  __bf16 v8bf;
typedef __attribute__((ext_vector_type(8)))  float  v8f;

__device__ __forceinline__ __bf16 f2bf(float f) {
    unsigned u = __builtin_bit_cast(unsigned, f);
    unsigned r = u + 0x7FFFu + ((u >> 16) & 1u);   // round-to-nearest-even
    unsigned short s = (unsigned short)(r >> 16);
    return __builtin_bit_cast(__bf16, s);
}

// ---------------------------------------------------------------------------
// Small utility kernels
// ---------------------------------------------------------------------------
__global__ void zero_f32(float* p, size_t cnt) {
    size_t i = (size_t)blockIdx.x * blockDim.x + threadIdx.x;
    size_t stride = (size_t)gridDim.x * blockDim.x;
    for (; i < cnt; i += stride) p[i] = 0.0f;
}

__global__ void zero_u32(unsigned* p, int cnt) {
    int i = blockIdx.x * blockDim.x + threadIdx.x;
    if (i < cnt) p[i] = 0u;
}

__global__ void init_edges(const int* __restrict__ s, const int* __restrict__ d,
                           int* __restrict__ so, int* __restrict__ d_o,
                           float* __restrict__ mo, int E) {
    int e = blockIdx.x * blockDim.x + threadIdx.x;
    if (e >= E) return;
    so[e]  = s[e];
    d_o[e] = d[e];
    mo[e]  = 1.0f;
}

// ---------------------------------------------------------------------------
// Prep: X (fp32 [n,din]) -> Xbf (bf16 [nPad,dinPad], zero padded)
//       W (fp32 [din,256]) -> Wt (bf16 [256,dinPad], transposed, zero padded)
// Padding removes ALL bounds checks from the WMMA GEMM inner loop.
// ---------------------------------------------------------------------------
__global__ void convert_x_bf16(const float* __restrict__ X, __bf16* __restrict__ Xbf,
                               int n, int din, int nPad, int dinPad) {
    size_t idx = (size_t)blockIdx.x * blockDim.x + threadIdx.x;
    size_t tot = (size_t)nPad * dinPad;
    if (idx >= tot) return;
    int r = (int)(idx / dinPad);
    int c = (int)(idx % dinPad);
    float v = (r < n && c < din) ? X[(size_t)r * din + c] : 0.0f;
    Xbf[idx] = f2bf(v);
}

__global__ void transpose_w_bf16(const float* __restrict__ W, __bf16* __restrict__ Wt,
                                 int din, int dinPad) {
    size_t idx = (size_t)blockIdx.x * blockDim.x + threadIdx.x;
    size_t tot = (size_t)HDIM * dinPad;
    if (idx >= tot) return;
    int c = (int)(idx / dinPad);      // N index (row of Wt)
    int r = (int)(idx % dinPad);      // K index
    float v = (r < din) ? W[(size_t)r * HDIM + c] : 0.0f;
    Wt[idx] = f2bf(v);
}

// ---------------------------------------------------------------------------
// WMMA GEMM:  XW[n,256] = X[n,din] @ W[din,256]   (bf16 inputs, f32 accum)
//   One wave computes a 16x128 strip: 8 N-tiles, v_wmma_f32_16x16x32_bf16,
//   K stepped by 32.  All loads are unconditional, 16B-aligned b128s.
//   Fragment layouts used here (h = lane/16, ml = lane%16):
//     A pairs 0..7: K = 16h+0..15 (contiguous)
//     B pairs 0..7: K = 16h+0..15 (contiguous)
//     C VGPR r -> (M = r + 8h, N = ml)
// ---------------------------------------------------------------------------
__global__ void gemm_xw_wmma(const __bf16* __restrict__ Xbf, const __bf16* __restrict__ Wt,
                             float* __restrict__ XW, int n, int dinPad) {
    const int lane = threadIdx.x & 31;
    const int wave = threadIdx.x >> 5;
    const int tilesM = (n + 15) >> 4;
    int wid = blockIdx.x * 8 + wave;
    if (wid >= tilesM * 2) return;
    const int tm     = wid >> 1;
    const int tnBase = (wid & 1) * NTPW;             // 0 or 8
    const int h  = lane >> 4;
    const int ml = lane & 15;

    const __bf16* arow = Xbf + (size_t)(tm * 16 + ml) * dinPad;

    v8f acc[NTPW];
#pragma unroll
    for (int t = 0; t < NTPW; ++t) acc[t] = (v8f){};

    for (int k0 = 0; k0 < dinPad; k0 += 32) {
        v16bf a = *(const v16bf*)(arow + k0 + 16 * h);
#pragma unroll
        for (int t = 0; t < NTPW; ++t) {
            const __bf16* bp = Wt + (size_t)((tnBase + t) * 16 + ml) * dinPad + k0 + 16 * h;
            v16bf b = *(const v16bf*)bp;
            acc[t] = __builtin_amdgcn_wmma_f32_16x16x32_bf16(
                /*neg_a=*/false, a, /*neg_b=*/false, b,
                /*c_mod=*/(short)0, acc[t], /*reuse_a=*/false, /*reuse_b=*/false);
        }
    }

#pragma unroll
    for (int t = 0; t < NTPW; ++t) {
#pragma unroll
        for (int r = 0; r < 8; ++r) {
            int row = tm * 16 + r + 8 * h;
            if (row < n) XW[(size_t)row * HDIM + (tnBase + t) * 16 + ml] = acc[t][r];
        }
    }
}

// ---------------------------------------------------------------------------
// GCNConv pieces
// ---------------------------------------------------------------------------
__global__ void init_deg(float* deg, int n) {     // 1.0 == implicit self loop
    int i = blockIdx.x * blockDim.x + threadIdx.x;
    if (i < n) deg[i] = 1.0f;
}

__global__ void scatter_deg(const int* __restrict__ dst, const float* __restrict__ mask,
                            float* __restrict__ deg, int E) {
    int e = blockIdx.x * blockDim.x + threadIdx.x;
    if (e >= E) return;
    float m = mask[e];
    if (m != 0.0f) atomicAdd(&deg[dst[e]], m);
}

__global__ void deg_rsqrt(const float* __restrict__ deg, float* __restrict__ dinv, int n) {
    int i = blockIdx.x * blockDim.x + threadIdx.x;
    if (i < n) dinv[i] = rsqrtf(deg[i]);
}

// one block (256 threads) per edge, thread f handles feature f (coalesced row)
__global__ void scatter_agg(const int* __restrict__ src, const int* __restrict__ dst,
                            const float* __restrict__ mask, const float* __restrict__ dinv,
                            const float* __restrict__ xw, float* __restrict__ agg) {
    int e = blockIdx.x;
    float m = mask[e];
    if (m == 0.0f) return;
    int s = src[e], d = dst[e];
    float coef = dinv[s] * dinv[d] * m;
    int f = threadIdx.x;
    atomicAdd(&agg[(size_t)d * HDIM + f], xw[(size_t)s * HDIM + f] * coef);
}

__global__ void finish_conv(const float* __restrict__ agg, const float* __restrict__ xw,
                            const float* __restrict__ dinv, const float* __restrict__ bias,
                            float* __restrict__ hout, int n) {
    size_t idx = (size_t)blockIdx.x * blockDim.x + threadIdx.x;
    if (idx >= (size_t)n * HDIM) return;
    int v = (int)(idx / HDIM);
    int f = (int)(idx % HDIM);
    float di = dinv[v];
    float val = agg[idx] + xw[idx] * di * di + bias[f];
    hout[idx] = fmaxf(val, 0.0f);
}

// ---------------------------------------------------------------------------
// SAGPool scoring: t[v] = h[v].Wrel ; score[v] = brel + h[v].Wroot
// ---------------------------------------------------------------------------
__global__ void node_dots(const float* __restrict__ hx, const float* __restrict__ Wrel,
                          const float* __restrict__ Wroot, const float* __restrict__ brel,
                          float* __restrict__ tvec, float* __restrict__ score, int n) {
    int lane = threadIdx.x & 31;
    int wave = threadIdx.x >> 5;
    int v = blockIdx.x * 8 + wave;
    if (v >= n) return;
    float s1 = 0.0f, s2 = 0.0f;
    for (int f = lane; f < HDIM; f += 32) {
        float hv = hx[(size_t)v * HDIM + f];
        s1 += hv * Wrel[f];
        s2 += hv * Wroot[f];
    }
#pragma unroll
    for (int off = 16; off > 0; off >>= 1) {
        s1 += __shfl_down(s1, off);
        s2 += __shfl_down(s2, off);
    }
    if (lane == 0) {
        tvec[v]  = s1;
        score[v] = s2 + brel[0];
    }
}

__global__ void scatter_score(const int* __restrict__ src, const int* __restrict__ dst,
                              const float* __restrict__ mask, const float* __restrict__ tvec,
                              float* __restrict__ score, int E) {
    int e = blockIdx.x * blockDim.x + threadIdx.x;
    if (e >= E) return;
    float m = mask[e];
    if (m != 0.0f) atomicAdd(&score[dst[e]], m * tvec[src[e]]);
}

// ---------------------------------------------------------------------------
// Exact top-k via 4-pass 8-bit radix select on order-transformed float bits.
// state[0]=prefix  state[1]=kRemain(->ties needed)  state[2]=#strictly-above
// state[4],state[5]=assignment counters
// ---------------------------------------------------------------------------
__device__ __forceinline__ unsigned orderKey(float s) {
    unsigned u = __builtin_bit_cast(unsigned, s);
    return (u & 0x80000000u) ? ~u : (u | 0x80000000u);
}

__global__ void topk_init_state(unsigned* state, int k) {
    if (threadIdx.x == 0) {
        state[0] = 0u;
        state[1] = (unsigned)k;
        state[2] = 0u;
        state[4] = 0u;
        state[5] = 0u;
    }
}

__global__ void topk_hist(const float* __restrict__ score, int n,
                          const unsigned* __restrict__ state, unsigned* __restrict__ hist,
                          int shift) {
    int i = blockIdx.x * blockDim.x + threadIdx.x;
    if (i >= n) return;
    unsigned key = orderKey(score[i]);
    if (shift < 24) {
        unsigned m = 0xFFFFFFFFu << (shift + 8);
        if ((key & m) != (state[0] & m)) return;
    }
    atomicAdd(&hist[(key >> shift) & 0xFFu], 1u);
}

__global__ void topk_select_bucket(unsigned* state, unsigned* hist, int shift) {
    if (threadIdx.x != 0) return;
    unsigned kRemain = state[1];
    unsigned cum = 0;
    int bsel = 0;
    for (int bkt = 255; bkt >= 0; --bkt) {
        unsigned c = hist[bkt];
        if (cum + c >= kRemain) { bsel = bkt; break; }
        cum += c;
    }
    state[0] |= ((unsigned)bsel) << shift;
    state[1]  = kRemain - cum;
    state[2] += cum;
    for (int j = 0; j < 256; ++j) hist[j] = 0u;   // clean for next pass
}

__global__ void topk_assign(const float* __restrict__ score, int n,
                            const unsigned* __restrict__ state, unsigned* __restrict__ ctr,
                            int* __restrict__ mapping, int* __restrict__ perm,
                            float* __restrict__ vals) {
    int v = blockIdx.x * blockDim.x + threadIdx.x;
    if (v >= n) return;
    unsigned key    = orderKey(score[v]);
    unsigned thr    = state[0];
    unsigned nAbove = state[2];
    unsigned nTies  = state[1];
    int newid = -1;
    if (key > thr) {
        newid = (int)atomicAdd(&ctr[0], 1u);
    } else if (key == thr) {
        unsigned p = atomicAdd(&ctr[1], 1u);
        if (p < nTies) newid = (int)(nAbove + p);
    }
    mapping[v] = newid;
    if (newid >= 0) {
        perm[newid] = v;
        vals[newid] = score[v];
    }
}

// ---------------------------------------------------------------------------
// Gate, rewire, readout
// ---------------------------------------------------------------------------
__global__ void gate_gather(const float* __restrict__ hx, const int* __restrict__ perm,
                            const float* __restrict__ vals, float* __restrict__ xnext, int k) {
    size_t idx = (size_t)blockIdx.x * blockDim.x + threadIdx.x;
    if (idx >= (size_t)k * HDIM) return;
    int j = (int)(idx / HDIM);
    int f = (int)(idx % HDIM);
    xnext[idx] = hx[(size_t)perm[j] * HDIM + f] * tanhf(vals[j]);
}

__global__ void rewire(const int* __restrict__ src, const int* __restrict__ dst,
                       const float* __restrict__ mask, const int* __restrict__ mapping,
                       int* __restrict__ nsrc, int* __restrict__ ndst,
                       float* __restrict__ nmask, int E) {
    int e = blockIdx.x * blockDim.x + threadIdx.x;
    if (e >= E) return;
    int a = mapping[src[e]];
    int b = mapping[dst[e]];
    nsrc[e]  = a < 0 ? 0 : a;
    ndst[e]  = b < 0 ? 0 : b;
    nmask[e] = (a >= 0 && b >= 0) ? mask[e] : 0.0f;
}

// one block per feature; accumulate mean || max into out (out pre-zeroed)
__global__ void readout_accum(const float* __restrict__ xp, int k, float* __restrict__ out) {
    __shared__ float ssum[256];
    __shared__ float smax[256];
    int f = blockIdx.x;
    int tid = threadIdx.x;
    float s = 0.0f, m = -3.402823e38f;
    for (int j = tid; j < k; j += blockDim.x) {
        float v = xp[(size_t)j * HDIM + f];
        s += v;
        m = fmaxf(m, v);
    }
    ssum[tid] = s;
    smax[tid] = m;
    __syncthreads();
    for (int off = 128; off > 0; off >>= 1) {
        if (tid < off) {
            ssum[tid] += ssum[tid + off];
            smax[tid]  = fmaxf(smax[tid], smax[tid + off]);
        }
        __syncthreads();
    }
    if (tid == 0) {
        out[f]        += ssum[0] / (float)k;
        out[HDIM + f] += smax[0];
    }
}

// ---------------------------------------------------------------------------
// Host-side pipeline
// ---------------------------------------------------------------------------
extern "C" void kernel_launch(void* const* d_in, const int* in_sizes, int n_in,
                              void* d_out, int out_size, void* d_ws, size_t ws_size,
                              hipStream_t stream) {
    (void)in_sizes; (void)n_in; (void)out_size; (void)ws_size;

    const float* x_in   = (const float*)d_in[0];
    const int*   src_in = (const int*)d_in[1];
    const int*   dst_in = (const int*)d_in[2];
    const float* convW[4]  = {(const float*)d_in[3],  (const float*)d_in[4],
                              (const float*)d_in[5],  (const float*)d_in[6]};
    const float* convB[4]  = {(const float*)d_in[7],  (const float*)d_in[8],
                              (const float*)d_in[9],  (const float*)d_in[10]};
    const float* pWrel[4]  = {(const float*)d_in[11], (const float*)d_in[12],
                              (const float*)d_in[13], (const float*)d_in[14]};
    const float* pbrel[4]  = {(const float*)d_in[15], (const float*)d_in[16],
                              (const float*)d_in[17], (const float*)d_in[18]};
    const float* pWroot[4] = {(const float*)d_in[19], (const float*)d_in[20],
                              (const float*)d_in[21], (const float*)d_in[22]};
    float* out = (float*)d_out;

    // ---- workspace carve-up (256B aligned) ----
    uint8_t* wp = (uint8_t*)d_ws;
    auto carve = [&](size_t bytes) -> void* {
        void* p = (void*)wp;
        wp += (bytes + 255) & ~(size_t)255;
        return p;
    };
    float*    xw      = (float*)carve((size_t)NNODE * HDIM * 4);
    float*    agg     = (float*)carve((size_t)NNODE * HDIM * 4);
    float*    hbuf    = (float*)carve((size_t)NNODE * HDIM * 4);
    float*    xnext   = (float*)carve((size_t)KSTAGE[1] * HDIM * 4);
    __bf16*   Xbf     = (__bf16*)carve((size_t)NNODE * DIN0P * 2);   // stage0 is max
    __bf16*   Wt      = (__bf16*)carve((size_t)HDIM * DIN0P * 2);
    float*    deg     = (float*)carve((size_t)NNODE * 4);
    float*    dinv    = (float*)carve((size_t)NNODE * 4);
    float*    tvec    = (float*)carve((size_t)NNODE * 4);
    float*    score   = (float*)carve((size_t)NNODE * 4);
    int*      mapping = (int*)carve((size_t)NNODE * 4);
    int*      perm    = (int*)carve((size_t)KSTAGE[1] * 4);
    float*    vals    = (float*)carve((size_t)KSTAGE[1] * 4);
    int*      srcA    = (int*)carve((size_t)NEDGE * 4);
    int*      dstA    = (int*)carve((size_t)NEDGE * 4);
    float*    mskA    = (float*)carve((size_t)NEDGE * 4);
    int*      srcB    = (int*)carve((size_t)NEDGE * 4);
    int*      dstB    = (int*)carve((size_t)NEDGE * 4);
    float*    mskB    = (float*)carve((size_t)NEDGE * 4);
    unsigned* hist    = (unsigned*)carve(256 * 4);
    unsigned* state   = (unsigned*)carve(8 * 4);

    const int T = 256;
    auto cdiv  = [](int a, int b) { return (a + b - 1) / b; };
    auto cdivz = [](size_t a, size_t b) { return (int)((a + b - 1) / b); };

    // ---- fresh output + initial edge buffers every call (deterministic) ----
    zero_f32<<<2, T, 0, stream>>>(out, (size_t)2 * HDIM);
    init_edges<<<cdiv(NEDGE, T), T, 0, stream>>>(src_in, dst_in, srcA, dstA, mskA, NEDGE);

    int*   srcC = srcA; int*   dstC = dstA; float* mskC = mskA;
    int*   srcN = srcB; int*   dstN = dstB; float* mskN = mskB;

    for (int i = 0; i < 4; ++i) {
        const int n      = KSTAGE[i];
        const int k      = KSTAGE[i + 1];
        const int din    = (i == 0) ? DIN0 : HDIM;
        const int dinPad = (i == 0) ? DIN0P : HDIM;
        const int nPad   = cdiv(n, 16) * 16;
        const float* X   = (i == 0) ? x_in : xnext;

        // -------- GCNConv: xw = X @ W via WMMA --------
        convert_x_bf16<<<cdivz((size_t)nPad * dinPad, T), T, 0, stream>>>(
            X, Xbf, n, din, nPad, dinPad);
        transpose_w_bf16<<<cdivz((size_t)HDIM * dinPad, T), T, 0, stream>>>(
            convW[i], Wt, din, dinPad);
        {
            int waves = (nPad / 16) * 2;             // 16x128 strip per wave
            gemm_xw_wmma<<<cdiv(waves, 8), T, 0, stream>>>(Xbf, Wt, xw, n, dinPad);
        }
        init_deg<<<cdiv(n, T), T, 0, stream>>>(deg, n);
        scatter_deg<<<cdiv(NEDGE, T), T, 0, stream>>>(dstC, mskC, deg, NEDGE);
        deg_rsqrt<<<cdiv(n, T), T, 0, stream>>>(deg, dinv, n);
        zero_f32<<<4096, T, 0, stream>>>(agg, (size_t)n * HDIM);
        scatter_agg<<<NEDGE, T, 0, stream>>>(srcC, dstC, mskC, dinv, xw, agg);
        finish_conv<<<cdiv(n * HDIM, T), T, 0, stream>>>(agg, xw, dinv, convB[i], hbuf, n);

        // -------- SAGPool scoring --------
        node_dots<<<cdiv(n, 8), T, 0, stream>>>(hbuf, pWrel[i], pWroot[i], pbrel[i],
                                                tvec, score, n);
        scatter_score<<<cdiv(NEDGE, T), T, 0, stream>>>(srcC, dstC, mskC, tvec, score, NEDGE);

        // -------- exact top-k radix select --------
        topk_init_state<<<1, 32, 0, stream>>>(state, k);
        zero_u32<<<1, 256, 0, stream>>>(hist, 256);
        for (int shift = 24; shift >= 0; shift -= 8) {
            topk_hist<<<cdiv(n, T), T, 0, stream>>>(score, n, state, hist, shift);
            topk_select_bucket<<<1, 32, 0, stream>>>(state, hist, shift);
        }
        topk_assign<<<cdiv(n, T), T, 0, stream>>>(score, n, state, state + 4,
                                                  mapping, perm, vals);

        // -------- gate + readout + rewire --------
        gate_gather<<<cdiv(k * HDIM, T), T, 0, stream>>>(hbuf, perm, vals, xnext, k);
        readout_accum<<<HDIM, T, 0, stream>>>(xnext, k, out);
        if (i < 3) {
            rewire<<<cdiv(NEDGE, T), T, 0, stream>>>(srcC, dstC, mskC, mapping,
                                                     srcN, dstN, mskN, NEDGE);
            int* ts; float* tm;
            ts = srcC; srcC = srcN; srcN = ts;
            ts = dstC; dstC = dstN; dstN = ts;
            tm = mskC; mskC = mskN; mskN = tm;
        }
    }
}
